// RecurrentLMGraves_68942815035729
// MI455X (gfx1250) — compile-verified, matrix operand order checked by reading, not run on hardware
//
#include <hip/hip_runtime.h>
#include <hip/hip_bf16.h>
#include <math.h>

#define B_  64
#define T_  512
#define V_  256
#define E_  512
#define H_  1024
#define L_  4
#define D_  512
#define K_IH (H_ + E_)   // 1536

typedef __attribute__((ext_vector_type(16))) __bf16 v16bf;
typedef __attribute__((ext_vector_type(8)))  float  v8f;

static __device__ __forceinline__ unsigned short f2bf(float f) {
  unsigned int u = __float_as_uint(f);
  return (unsigned short)((u + 0x7FFFu + ((u >> 16) & 1u)) >> 16);
}

// Read one 16x32 bf16 WMMA fragment given this lane's pre-offset pointer.
// Layout per CDNA5 ISA 7.12.2 (16-bit matrices): lane l<16 holds row l,
// K {0..7, 16..23}; lane l>=16 holds row l-16, K {8..15, 24..31}.
// Caller bakes (row + lane%16)*ld + (lane/16)*8 into p; the two 16-byte
// chunks sit at p and p+16 elements.
static __device__ __forceinline__ v16bf frag_at(const unsigned short* __restrict__ p) {
  union { uint4 q[2]; v16bf v; } u;
  u.q[0] = *(const uint4*)p;
  u.q[1] = *(const uint4*)(p + 16);
  return u.v;
}

static __device__ __forceinline__ v8f wmma_bf16(v16bf a, v16bf b, v8f c) {
  return __builtin_amdgcn_wmma_f32_16x16x32_bf16(false, a, false, b, (short)0, c, false, false);
}

// ---------------- prep kernels ----------------

__global__ void k_cvt(const float* __restrict__ in, unsigned short* __restrict__ out, size_t n) {
  size_t i = (size_t)blockIdx.x * blockDim.x + threadIdx.x;
  size_t st = (size_t)gridDim.x * blockDim.x;
  for (; i < n; i += st) out[i] = f2bf(in[i]);
}

// in: [R,C] row-major fp32 -> out: [C,R] row-major bf16 (transpose+convert)
__global__ void k_tcvt(const float* __restrict__ in, unsigned short* __restrict__ out, int R, int C) {
  size_t n = (size_t)R * C;
  size_t i = (size_t)blockIdx.x * blockDim.x + threadIdx.x;
  size_t st = (size_t)gridDim.x * blockDim.x;
  for (; i < n; i += st) {
    int r = (int)(i / C), c = (int)(i % C);
    out[(size_t)c * R + r] = f2bf(in[i]);
  }
}

// e[t][b][k] = bf16(embed[tok[b][t]][k])
__global__ void k_embed(const int* __restrict__ tok, const float* __restrict__ embed,
                        unsigned short* __restrict__ e) {
  size_t n = (size_t)T_ * B_ * E_;
  size_t i = (size_t)blockIdx.x * blockDim.x + threadIdx.x;
  size_t st = (size_t)gridDim.x * blockDim.x;
  for (; i < n; i += st) {
    int t = (int)(i / ((size_t)B_ * E_));
    int rem = (int)(i % ((size_t)B_ * E_));
    int b = rem / E_, k = rem % E_;
    e[i] = f2bf(embed[(size_t)tok[b * T_ + t] * E_ + k]);
  }
}

// h[l][b][:] = h0[l][:]  (fp32 + bf16 copies)
__global__ void k_hinit(const float* __restrict__ h0, float* __restrict__ hf,
                        unsigned short* __restrict__ hb) {
  size_t n = (size_t)L_ * B_ * H_;
  size_t i = (size_t)blockIdx.x * blockDim.x + threadIdx.x;
  size_t st = (size_t)gridDim.x * blockDim.x;
  for (; i < n; i += st) {
    int l = (int)(i / ((size_t)B_ * H_));
    int k = (int)(i % H_);
    float v = h0[l * H_ + k];
    hf[i] = v;
    hb[i] = f2bf(v);
  }
}

// ---------------- persistent GRU recurrence ----------------
// grid = 128 WGs: layer = blk>>5, H-slice (32 cols) = blk&31. 256 thr = 8 waves,
// wave w: Mtile = w&3 (16 of the 64 batch rows), Nt = w>>2 (16 of the 32 slice cols).
// Each wave owns 6 f32 accumulators: {ir,iz,in} x {gi via W_ih, gh via W_hh}.
// bf16 weights (~63 MB total) stay resident in the 192 MB L2 across all 512 steps.
__global__ void __launch_bounds__(256, 1)
k_gru(const unsigned short* __restrict__ Wih, const unsigned short* __restrict__ Whh,
      const float* __restrict__ bih, const float* __restrict__ bhh,
      const unsigned short* __restrict__ e,
      float* hf0, float* hf1, unsigned short* hb0, unsigned short* hb1,
      unsigned short* __restrict__ tops, unsigned int* __restrict__ ctr)
{
  const int layer = blockIdx.x >> 5;
  const int slice = blockIdx.x & 31;
  const int w     = threadIdx.x >> 5;
  const int lane  = threadIdx.x & 31;
  const int Mtile = w & 3;
  const int Nt    = w >> 2;
  const int nbase = slice * 32 + Nt * 16;          // H-column tile base
  const int rlane = lane & 15;                     // fragment row within tile
  const int koff  = (lane >> 4) << 3;              // fragment K sub-offset
  const int ncol  = nbase + rlane;                 // this lane's H column
  const int row0  = layer * B_ + Mtile * 16;       // row base in h buffers [L*B, H]

  // Per-lane weight fragment base pointers: invariant across t; only += 32/kt.
  const unsigned short* Wih_l = Wih + (size_t)layer * 3 * H_ * K_IH;
  const unsigned short* Whh_l = Whh + (size_t)layer * 3 * H_ * H_;
  const unsigned short* pbr0 = Wih_l + (size_t)(0 * H_ + nbase + rlane) * K_IH + koff;
  const unsigned short* pbz0 = Wih_l + (size_t)(1 * H_ + nbase + rlane) * K_IH + koff;
  const unsigned short* pbn0 = Wih_l + (size_t)(2 * H_ + nbase + rlane) * K_IH + koff;
  const unsigned short* pcr0 = Whh_l + (size_t)(0 * H_ + nbase + rlane) * H_ + koff;
  const unsigned short* pcz0 = Whh_l + (size_t)(1 * H_ + nbase + rlane) * H_ + koff;
  const unsigned short* pcn0 = Whh_l + (size_t)(2 * H_ + nbase + rlane) * H_ + koff;

  const float bihr = bih[layer * 3 * H_ + 0 * H_ + ncol];
  const float bihz = bih[layer * 3 * H_ + 1 * H_ + ncol];
  const float bihn = bih[layer * 3 * H_ + 2 * H_ + ncol];
  const float bhhr = bhh[layer * 3 * H_ + 0 * H_ + ncol];
  const float bhhz = bhh[layer * 3 * H_ + 1 * H_ + ncol];
  const float bhhn = bhh[layer * 3 * H_ + 2 * H_ + ncol];

  const size_t aoff  = (size_t)(row0 + rlane) * H_ + koff;           // into h bf16
  const size_t eoff  = (size_t)(Mtile * 16 + rlane) * E_ + koff;     // into e_t
  const unsigned int nwg = gridDim.x;

  for (int t = 0; t < T_; ++t) {
    const float*          hf_cur = (t & 1) ? hf1 : hf0;
    float*                hf_nxt = (t & 1) ? hf0 : hf1;
    const unsigned short* hb_cur = (t & 1) ? hb1 : hb0;
    unsigned short*       hb_nxt = (t & 1) ? hb0 : hb1;

    v8f acc_ir = {0,0,0,0,0,0,0,0}, acc_iz = {0,0,0,0,0,0,0,0}, acc_in = {0,0,0,0,0,0,0,0};
    v8f acc_hr = {0,0,0,0,0,0,0,0}, acc_hz = {0,0,0,0,0,0,0,0}, acc_hn = {0,0,0,0,0,0,0,0};

    // ---- K region 1: x = h (K 0..1023): gi += h*Wih, gh += h*Whh ----
    {
      const unsigned short* pa  = hb_cur + aoff;
      const unsigned short* pbr = pbr0;  const unsigned short* pbz = pbz0;
      const unsigned short* pbn = pbn0;  const unsigned short* pcr = pcr0;
      const unsigned short* pcz = pcz0;  const unsigned short* pcn = pcn0;
      #pragma unroll 2
      for (int kt = 0; kt < H_ / 32; ++kt) {
        v16bf a  = frag_at(pa);   pa  += 32;
        v16bf br = frag_at(pbr);  pbr += 32;
        v16bf bz = frag_at(pbz);  pbz += 32;
        v16bf bn = frag_at(pbn);  pbn += 32;
        v16bf cr = frag_at(pcr);  pcr += 32;
        v16bf cz = frag_at(pcz);  pcz += 32;
        v16bf cn = frag_at(pcn);  pcn += 32;
        acc_ir = wmma_bf16(a, br, acc_ir);
        acc_iz = wmma_bf16(a, bz, acc_iz);
        acc_in = wmma_bf16(a, bn, acc_in);
        acc_hr = wmma_bf16(a, cr, acc_hr);
        acc_hz = wmma_bf16(a, cz, acc_hz);
        acc_hn = wmma_bf16(a, cn, acc_hn);
      }
    }
    // ---- K region 2: x = e_t (K 1024..1535): gi += e*Wih ----
    {
      const unsigned short* pa  = e + (size_t)t * B_ * E_ + eoff;
      const unsigned short* pbr = pbr0 + H_;
      const unsigned short* pbz = pbz0 + H_;
      const unsigned short* pbn = pbn0 + H_;
      #pragma unroll 2
      for (int kt = 0; kt < E_ / 32; ++kt) {
        v16bf a  = frag_at(pa);   pa  += 32;
        v16bf br = frag_at(pbr);  pbr += 32;
        v16bf bz = frag_at(pbz);  pbz += 32;
        v16bf bn = frag_at(pbn);  pbn += 32;
        acc_ir = wmma_bf16(a, br, acc_ir);
        acc_iz = wmma_bf16(a, bz, acc_iz);
        acc_in = wmma_bf16(a, bn, acc_in);
      }
    }

    // GRU gate update (C-layout: vgpr j, lanes<16 -> m=j, lanes>=16 -> m=j+8)
    #pragma unroll
    for (int j = 0; j < 8; ++j) {
      int b = Mtile * 16 + j + ((lane >> 4) << 3);
      size_t idx = (size_t)(layer * B_ + b) * H_ + ncol;
      float hprev = hf_cur[idx];
      float rg = 1.f / (1.f + __expf(-(acc_ir[j] + bihr + acc_hr[j] + bhhr)));
      float zg = 1.f / (1.f + __expf(-(acc_iz[j] + bihz + acc_hz[j] + bhhz)));
      float ng = tanhf(acc_in[j] + bihn + rg * (acc_hn[j] + bhhn));
      float hnew = (1.f - zg) * ng + zg * hprev;
      hf_nxt[idx] = hnew;
      unsigned short hbv = f2bf(hnew);
      hb_nxt[idx] = hbv;
      if (layer == L_ - 1)
        tops[((size_t)t * B_ + b) * H_ + ncol] = hbv;   // top layer feeds decoder
    }

    // device-wide step barrier (monotonic counter, reset each launch)
    __syncthreads();
    if (threadIdx.x == 0) {
      __threadfence();
      atomicAdd(ctr, 1u);
      unsigned int target = nwg * (unsigned int)(t + 1);
      while (__hip_atomic_load(ctr, __ATOMIC_ACQUIRE, __HIP_MEMORY_SCOPE_AGENT) < target) {
        __builtin_amdgcn_s_sleep(8);
      }
      __threadfence();
    }
    __syncthreads();
  }
}

// ---------------- decoder GEMM 1: hdec = relu(tops @ W1 + b1) ----------------
// C [T*B=32768, D=512] ; A = tops [32768, 1024] bf16 ; B = W1T [512, 1024] bf16
__global__ void __launch_bounds__(256)
k_dec1(const unsigned short* __restrict__ tops, const unsigned short* __restrict__ W1T,
       const float* __restrict__ b1, unsigned short* __restrict__ hdec)
{
  const int gx = blockIdx.x;            // 512 row blocks of 64
  const int gy = blockIdx.y;            // 8 col blocks of 64
  const int w = threadIdx.x >> 5, lane = threadIdx.x & 31;
  const int Mtile = w & 3, nh = w >> 2;
  const int rlane = lane & 15, koff = (lane >> 4) << 3;
  const int row0 = gx * 64 + Mtile * 16;
  const int n0 = gy * 64 + nh * 32;
  const unsigned short* pa  = tops + (size_t)(row0 + rlane) * H_ + koff;
  const unsigned short* pb0 = W1T + (size_t)(n0 + rlane) * H_ + koff;
  const unsigned short* pb1 = W1T + (size_t)(n0 + 16 + rlane) * H_ + koff;
  v8f acc0 = {0,0,0,0,0,0,0,0}, acc1 = {0,0,0,0,0,0,0,0};
  #pragma unroll 2
  for (int kt = 0; kt < H_ / 32; ++kt) {
    v16bf a  = frag_at(pa);   pa  += 32;
    v16bf b0 = frag_at(pb0);  pb0 += 32;
    v16bf b1v= frag_at(pb1);  pb1 += 32;
    acc0 = wmma_bf16(a, b0, acc0);
    acc1 = wmma_bf16(a, b1v, acc1);
  }
  #pragma unroll
  for (int c = 0; c < 2; ++c) {
    v8f acc = c ? acc1 : acc0;
    int n = n0 + c * 16 + rlane;
    float bias = b1[n];
    #pragma unroll
    for (int j = 0; j < 8; ++j) {
      int m = row0 + j + ((lane >> 4) << 3);
      float v = acc[j] + bias;
      v = v > 0.f ? v : 0.f;
      hdec[(size_t)m * D_ + n] = f2bf(v);
    }
  }
}

// ---------------- decoder GEMM 2: logits = hdec @ W2 + b2 -> out [B,T,V] ----------------
__global__ void __launch_bounds__(256)
k_dec2(const unsigned short* __restrict__ hdec, const unsigned short* __restrict__ W2T,
       const float* __restrict__ b2, float* __restrict__ out)
{
  const int gx = blockIdx.x;            // 512 row blocks of 64
  const int gy = blockIdx.y;            // 4 col blocks of 64
  const int w = threadIdx.x >> 5, lane = threadIdx.x & 31;
  const int Mtile = w & 3, nh = w >> 2;
  const int rlane = lane & 15, koff = (lane >> 4) << 3;
  const int row0 = gx * 64 + Mtile * 16;
  const int n0 = gy * 64 + nh * 32;
  const unsigned short* pa  = hdec + (size_t)(row0 + rlane) * D_ + koff;
  const unsigned short* pb0 = W2T + (size_t)(n0 + rlane) * D_ + koff;
  const unsigned short* pb1 = W2T + (size_t)(n0 + 16 + rlane) * D_ + koff;
  v8f acc0 = {0,0,0,0,0,0,0,0}, acc1 = {0,0,0,0,0,0,0,0};
  #pragma unroll 2
  for (int kt = 0; kt < D_ / 32; ++kt) {
    v16bf a  = frag_at(pa);   pa  += 32;
    v16bf b0 = frag_at(pb0);  pb0 += 32;
    v16bf b1v= frag_at(pb1);  pb1 += 32;
    acc0 = wmma_bf16(a, b0, acc0);
    acc1 = wmma_bf16(a, b1v, acc1);
  }
  #pragma unroll
  for (int c = 0; c < 2; ++c) {
    v8f acc = c ? acc1 : acc0;
    int n = n0 + c * 16 + rlane;
    float bias = b2[n];
    #pragma unroll
    for (int j = 0; j < 8; ++j) {
      int m = row0 + j + ((lane >> 4) << 3);   // m = t*B + b
      int b = m % B_, t = m / B_;
      out[((size_t)b * T_ + t) * V_ + n] = acc[j] + bias;
    }
  }
}

// ---------------- host ----------------

extern "C" void kernel_launch(void* const* d_in, const int* in_sizes, int n_in,
                              void* d_out, int out_size, void* d_ws, size_t ws_size,
                              hipStream_t stream) {
  const int*   tok   = (const int*)d_in[0];
  const float* embed = (const float*)d_in[1];
  const float* W_ih  = (const float*)d_in[2];
  const float* W_hh  = (const float*)d_in[3];
  const float* b_ih  = (const float*)d_in[4];
  const float* b_hh  = (const float*)d_in[5];
  const float* h0    = (const float*)d_in[6];
  const float* dW1   = (const float*)d_in[7];
  const float* db1   = (const float*)d_in[8];
  const float* dW2   = (const float*)d_in[9];
  const float* db2   = (const float*)d_in[10];
  float* out = (float*)d_out;

  char* p = (char*)d_ws;
  auto alloc = [&](size_t bytes) -> char* {
    char* r = p; p += (bytes + 255) & ~(size_t)255; return r;
  };
  unsigned int*   ctr  = (unsigned int*)  alloc(256);
  unsigned short* wih  = (unsigned short*)alloc((size_t)L_ * 3 * H_ * K_IH * 2);
  unsigned short* whh  = (unsigned short*)alloc((size_t)L_ * 3 * H_ * H_ * 2);
  unsigned short* w1t  = (unsigned short*)alloc((size_t)D_ * H_ * 2);
  unsigned short* w2t  = (unsigned short*)alloc((size_t)V_ * D_ * 2);
  unsigned short* ebf  = (unsigned short*)alloc((size_t)T_ * B_ * E_ * 2);
  float*          hf0  = (float*)         alloc((size_t)L_ * B_ * H_ * 4);
  float*          hf1  = (float*)         alloc((size_t)L_ * B_ * H_ * 4);
  unsigned short* hb0  = (unsigned short*)alloc((size_t)L_ * B_ * H_ * 2);
  unsigned short* hb1  = (unsigned short*)alloc((size_t)L_ * B_ * H_ * 2);
  unsigned short* tops = (unsigned short*)alloc((size_t)T_ * B_ * H_ * 2);
  unsigned short* hdec = (unsigned short*)alloc((size_t)T_ * B_ * D_ * 2);

  hipMemsetAsync(ctr, 0, 256, stream);
  k_cvt  <<<2048, 256, 0, stream>>>(W_ih, wih, (size_t)L_ * 3 * H_ * K_IH);
  k_cvt  <<<2048, 256, 0, stream>>>(W_hh, whh, (size_t)L_ * 3 * H_ * H_);
  k_tcvt <<<1024, 256, 0, stream>>>(dW1, w1t, H_, D_);
  k_tcvt <<<256,  256, 0, stream>>>(dW2, w2t, D_, V_);
  k_embed<<<4096, 256, 0, stream>>>(tok, embed, ebf);
  k_hinit<<<1024, 256, 0, stream>>>(h0, hf0, hb0);
  k_gru  <<<L_ * 32, 256, 0, stream>>>(wih, whh, b_ih, b_hh, ebf,
                                       hf0, hf1, hb0, hb1, tops, ctr);
  k_dec1 <<<dim3((T_ * B_) / 64, D_ / 64), 256, 0, stream>>>(tops, w1t, db1, hdec);
  k_dec2 <<<dim3((T_ * B_) / 64, V_ / 64), 256, 0, stream>>>(hdec, w2t, db2, out);
}
